// GNFRenderer_89215060672774
// MI455X (gfx1250) — matrix operand order repaired and checked by reading, not compiled, not run on hardware
//
#include <hip/hip_runtime.h>
#include <hip/hip_bf16.h>
#include <hip/hip_fp16.h>

typedef __attribute__((ext_vector_type(16))) _Float16 v16h;
typedef __attribute__((ext_vector_type(2)))  __fp16   hf2;
typedef __attribute__((ext_vector_type(8)))  float    v8f;

#define NR 4096
#define NS 512
#define NPTS (NR * NS)           // 2,097,152 points
#define NTILES (NPTS / 16)       // 131072 16-point tiles
#define BLOCKS_A 2048
#define TILES_PER_WAVE (NTILES / (BLOCKS_A * 8))   // 8

__device__ __forceinline__ v8f wmma_f16(v16h a, v16h b, v8f c) {
    // D = A(16x32 f16) * B(32x16 f16) + C(16x16 f32)
    return __builtin_amdgcn_wmma_f32_16x16x32_f16(
        false, a, false, b, (short)0, c, false, false);
}

// A-fragment of W^T (rows = output dims of M-block mb, Kdim = input dims)
// from LDS row-major Wt[outdim][indim]; per-lane reads are 2x16B contiguous.
__device__ __forceinline__ v16h load_a_w(const _Float16* Wt, int Kdim, int mb, int kb) {
    int lane = threadIdx.x & 31;
    int m = lane & 15, h = lane >> 4;
    const _Float16* rowp = Wt + (mb * 16 + m) * Kdim + kb * 32;
    v16h a;
#pragma unroll
    for (int v = 0; v < 8; ++v) {
        int k = h * 16 + v * 2;
        a[2 * v]     = rowp[k];
        a[2 * v + 1] = rowp[k + 1];
    }
    return a;
}

// C init with row-bias: C[row][pt] = b[mb*16 + r + 8h] (contiguous 32B LDS load).
__device__ __forceinline__ v8f bias_c(const float* sB, int mb) {
    int h = (threadIdx.x >> 4) & 1;
    const float* bp = sB + mb * 16 + 8 * h;
    v8f c;
#pragma unroll
    for (int r = 0; r < 8; ++r) c[r] = bp[r];
    return c;
}

// pack two f32 -> f16 pair (v_cvt_pk_rtz_f16_f32), ReLU in packed f16
// (v_pk_max_num_f16). relu commutes with the monotone f32->f16 rounding.
__device__ __forceinline__ unsigned pk_relu_f16(float x, float y) {
    hf2 v = __builtin_amdgcn_cvt_pkrtz(x, y);
    hf2 z = { (__fp16)0.0f, (__fp16)0.0f };
    hf2 r = __builtin_elementwise_max(v, z);
    union { hf2 v; unsigned u; } t;
    t.v = r;
    return t.u;
}

// D->B re-layout fully in registers: combine raw WMMA accumulators of M-blocks
// (even=cA, odd=cB) into the relu'd f16 B-frag for the matching 32-row K-chunk.
// Values are packed (with ReLU) first, so only 4 xor-16 exchanges per call.
__device__ __forceinline__ v16h d_to_b_relu(const v8f& cA, const v8f& cB) {
    int h = (threadIdx.x >> 4) & 1;
    unsigned pa[4], pb[4];
#pragma unroll
    for (int j = 0; j < 4; ++j) {
        pa[j] = pk_relu_f16(cA[2 * j], cA[2 * j + 1]);
        pb[j] = pk_relu_f16(cB[2 * j], cB[2 * j + 1]);
    }
    union { v16h v; unsigned u[8]; } b;
#pragma unroll
    for (int j = 0; j < 4; ++j) {
        unsigned send  = h ? pa[j] : pb[j];                   // partner's missing half
        unsigned other = (unsigned)__shfl_xor((int)send, 16, 32);
        b.u[j]     = h ? other : pa[j];                       // chunk-local rows 0..7
        b.u[4 + j] = h ? pb[j] : other;                       // chunk-local rows 8..15
    }
    return b.v;
}

__device__ __forceinline__ float sigmoidf_(float x) {
    return 1.0f / (1.0f + __expf(-x));
}

// ---------------------------------------------------------------------------
// Kernel A: transposed MLP (Y = W^T X^T) via WMMA + alpha-grid trilinear mask.
// 256 threads = 8 waves; each wave loops over TILES_PER_WAVE 16-point tiles,
// no inter-layer LDS traffic; weights live in registers after first iteration.
// ---------------------------------------------------------------------------
__global__ __launch_bounds__(256) void mlp_alpha_kernel(
    const float* __restrict__ xyz,      // [NPTS,3]
    const float* __restrict__ features, // [NPTS,32]
    const float* __restrict__ dists,    // [NPTS]
    const float* __restrict__ vol,      // [128,128,128]
    const float* __restrict__ W0, const float* __restrict__ b0,   // [32,64],[64]
    const float* __restrict__ W1, const float* __restrict__ b1,   // [64,64],[64]
    const float* __restrict__ W2, const float* __restrict__ b2,   // [64,49],[49]
    float4* __restrict__ ws)            // [NPTS] {alpha, r, g, b}
{
    __shared__ _Float16 sW0t[64 * 32];   // W0^T: [outdim=64][indim=32]
    __shared__ _Float16 sW1t[64 * 64];   // W1^T: [outdim=64][indim=64]
    __shared__ _Float16 sW2t[16 * 64];   // W2^T rows 0..3 (rest 0): [16][64]
    __shared__ float    sB0[64], sB1[64], sB2p[16];

    const int tid = threadIdx.x;

    // --- cooperative weight conversion (f32 -> f16, transposed), once per block ---
    for (int i = tid; i < 64 * 32; i += 256) {          // sW0t[o*32+k] = W0[k][o]
        int o = i >> 5, k = i & 31;
        sW0t[i] = (_Float16)W0[k * 64 + o];
    }
    for (int i = tid; i < 64 * 64; i += 256) {          // sW1t[o*64+k] = W1[k][o]
        int o = i >> 6, k = i & 63;
        sW1t[i] = (_Float16)W1[k * 64 + o];
    }
    for (int i = tid; i < 16 * 64; i += 256) {          // sW2t[o*64+k] = W2[k][o] (o<4)
        int o = i >> 6, k = i & 63;
        sW2t[i] = (o < 4) ? (_Float16)W2[k * 49 + o] : (_Float16)0.0f;
    }
    if (tid < 64) { sB0[tid] = b0[tid]; sB1[tid] = b1[tid]; }
    if (tid < 16) { sB2p[tid] = (tid < 4) ? b2[tid] : 0.0f; }
    __syncthreads();

    const int wave = tid >> 5;
    const int lane = tid & 31;
    const int n = lane & 15, h = lane >> 4;
    const long long tile0 = (long long)(blockIdx.x * 8 + wave) * TILES_PER_WAVE;

#pragma unroll 1
    for (int t = 0; t < TILES_PER_WAVE; ++t) {
        const long long p0 = (tile0 + t) * 16;

        // B-frag of X^T: lane (n,h) holds point n, feature channels h*16+2v(+1);
        // 16 contiguous f32 per lane -> 4x b128 global loads.
        v16h bx;
        {
            const float* frow = features + (p0 + n) * 32 + h * 16;
#pragma unroll
            for (int v = 0; v < 8; ++v) {
                bx[2 * v]     = (_Float16)frow[2 * v];
                bx[2 * v + 1] = (_Float16)frow[2 * v + 1];
            }
        }

        // layer 0: Y0 = relu(W0^T X^T + b0), 4 M-blocks
        v8f c0 = wmma_f16(load_a_w(sW0t, 32, 0, 0), bx, bias_c(sB0, 0));
        v8f c1 = wmma_f16(load_a_w(sW0t, 32, 1, 0), bx, bias_c(sB0, 1));
        v8f c2 = wmma_f16(load_a_w(sW0t, 32, 2, 0), bx, bias_c(sB0, 2));
        v8f c3 = wmma_f16(load_a_w(sW0t, 32, 3, 0), bx, bias_c(sB0, 3));
        v16h by0 = d_to_b_relu(c0, c1);   // Y0 rows  0..31
        v16h by1 = d_to_b_relu(c2, c3);   // Y0 rows 32..63

        // layer 1: Y1 = relu(W1^T Y0 + b1)
        c0 = bias_c(sB1, 0);
        c0 = wmma_f16(load_a_w(sW1t, 64, 0, 0), by0, c0);
        c0 = wmma_f16(load_a_w(sW1t, 64, 0, 1), by1, c0);
        c1 = bias_c(sB1, 1);
        c1 = wmma_f16(load_a_w(sW1t, 64, 1, 0), by0, c1);
        c1 = wmma_f16(load_a_w(sW1t, 64, 1, 1), by1, c1);
        c2 = bias_c(sB1, 2);
        c2 = wmma_f16(load_a_w(sW1t, 64, 2, 0), by0, c2);
        c2 = wmma_f16(load_a_w(sW1t, 64, 2, 1), by1, c2);
        c3 = bias_c(sB1, 3);
        c3 = wmma_f16(load_a_w(sW1t, 64, 3, 0), by0, c3);
        c3 = wmma_f16(load_a_w(sW1t, 64, 3, 1), by1, c3);
        by0 = d_to_b_relu(c0, c1);
        by1 = d_to_b_relu(c2, c3);

        // layer 2: out^T = W2p^T Y1 + b2 (only output dims 0..3 meaningful)
        v8f o = bias_c(sB2p, 0);
        o = wmma_f16(load_a_w(sW2t, 64, 0, 0), by0, o);
        o = wmma_f16(load_a_w(sW2t, 64, 0, 1), by1, o);

        // epilogue: lane (n, h=0) holds point n with outdims 0..3 in o[0..3]
        if (h == 0) {
            long long p = p0 + n;
            float o0 = o[0], o1 = o[1], o2 = o[2], o3 = o[3];

            float px = xyz[p * 3 + 0], py = xyz[p * 3 + 1], pz = xyz[p * 3 + 2];
            // ix = clip(((p+1.5)*(2/3)-1 + 1)*0.5*127) = clip((p+1.5)*127/3)
            float ix = fminf(fmaxf((px + 1.5f) * (127.0f / 3.0f), 0.0f), 127.0f);
            float iy = fminf(fmaxf((py + 1.5f) * (127.0f / 3.0f), 0.0f), 127.0f);
            float iz = fminf(fmaxf((pz + 1.5f) * (127.0f / 3.0f), 0.0f), 127.0f);
            int x0 = (int)floorf(ix), y0 = (int)floorf(iy), z0 = (int)floorf(iz);
            int x1 = min(x0 + 1, 127), y1 = min(y0 + 1, 127), z1 = min(z0 + 1, 127);
            float wx = ix - (float)x0, wy = iy - (float)y0, wz = iz - (float)z0;
            const float* vz0 = vol + z0 * 16384;
            const float* vz1 = vol + z1 * 16384;
            float c000 = vz0[y0 * 128 + x0], c001 = vz0[y0 * 128 + x1];
            float c010 = vz0[y1 * 128 + x0], c011 = vz0[y1 * 128 + x1];
            float c100 = vz1[y0 * 128 + x0], c101 = vz1[y0 * 128 + x1];
            float c110 = vz1[y1 * 128 + x0], c111 = vz1[y1 * 128 + x1];
            float c00 = c000 + (c001 - c000) * wx;
            float c01 = c010 + (c011 - c010) * wx;
            float c10 = c100 + (c101 - c100) * wx;
            float c11 = c110 + (c111 - c110) * wx;
            float cz0 = c00 + (c01 - c00) * wy;
            float cz1 = c10 + (c11 - c10) * wy;
            float aval = cz0 + (cz1 - cz0) * wz;

            float sigma = (aval > 0.5f) ? fmaxf(o0, 0.0f) : 0.0f;
            float alpha = 1.0f - __expf(-sigma * dists[p]);
            ws[p] = make_float4(alpha, sigmoidf_(o1), sigmoidf_(o2), sigmoidf_(o3));
        }
    }
}

// ---------------------------------------------------------------------------
// Kernel B: per-ray alpha compositing. One wave32 per ray, shfl scan.
// ---------------------------------------------------------------------------
__global__ __launch_bounds__(256) void composite_kernel(
    const float4* __restrict__ ws, float* __restrict__ out)
{
    const int ray  = (int)((blockIdx.x * 256 + threadIdx.x) >> 5);
    const int lane = threadIdx.x & 31;
    const float4* row = ws + (size_t)ray * NS;

    float T = 1.0f, acc = 0.0f, cr = 0.0f, cg = 0.0f, cb = 0.0f;
#pragma unroll 1
    for (int ch = 0; ch < NS / 32; ++ch) {
        float4 v = row[ch * 32 + lane];
        float a = v.x;
        float t = 1.0f - a + 1e-10f;
        // inclusive prefix product across the wave
        float pr = t;
#pragma unroll
        for (int off = 1; off < 32; off <<= 1) {
            float q = __shfl_up(pr, off, 32);
            if (lane >= off) pr *= q;
        }
        float excl = __shfl_up(pr, 1, 32);
        if (lane == 0) excl = 1.0f;
        float w = a * T * excl;
        acc += w;
        cr += w * v.y; cg += w * v.z; cb += w * v.w;
        T *= __shfl(pr, 31, 32);   // carry transmittance to next chunk
    }
#pragma unroll
    for (int off = 16; off >= 1; off >>= 1) {
        acc += __shfl_down(acc, off, 32);
        cr  += __shfl_down(cr,  off, 32);
        cg  += __shfl_down(cg,  off, 32);
        cb  += __shfl_down(cb,  off, 32);
    }
    if (lane == 0) {
        float bg = 1.0f - acc;
        out[ray * 3 + 0] = cr + bg;
        out[ray * 3 + 1] = cg + bg;
        out[ray * 3 + 2] = cb + bg;
    }
}

extern "C" void kernel_launch(void* const* d_in, const int* in_sizes, int n_in,
                              void* d_out, int out_size, void* d_ws, size_t ws_size,
                              hipStream_t stream) {
    const float* xyz      = (const float*)d_in[0];
    const float* features = (const float*)d_in[1];
    const float* dists    = (const float*)d_in[2];
    const float* vol      = (const float*)d_in[3];
    const float* W0 = (const float*)d_in[4];
    const float* b0 = (const float*)d_in[5];
    const float* W1 = (const float*)d_in[6];
    const float* b1 = (const float*)d_in[7];
    const float* W2 = (const float*)d_in[8];
    const float* b2 = (const float*)d_in[9];
    float4* ws = (float4*)d_ws;   // NPTS * 16B = 32 MB scratch
    float*  out = (float*)d_out;  // [4096,3]

    mlp_alpha_kernel<<<BLOCKS_A, 256, 0, stream>>>(
        xyz, features, dists, vol, W0, b0, W1, b1, W2, b2, ws);
    // 4096 rays * 32 lanes / 256 = 512 blocks
    composite_kernel<<<NR * 32 / 256, 256, 0, stream>>>(ws, out);
}
